// CausalMHSA_3882650437127
// MI455X (gfx1250) — matrix operand order repaired and checked by reading, not compiled, not run on hardware
//
#include <hip/hip_runtime.h>
#include <math.h>

typedef __attribute__((ext_vector_type(16))) __bf16 v16bf;
typedef __attribute__((ext_vector_type(8)))  float  v8f;

#define DEV __device__ __forceinline__

constexpr int Bb = 2, Ss = 2048, Dd = 1024, Hh = 16, DK = 64;
constexpr float ATTN_SCALE = 0.125f;  // 1/sqrt(64), power of two -> exact in bf16

// ---------------- WMMA fragment loaders (contiguous k order, wave32) -------
// A (16xK=32 bf16): lane row M = lane&15; lanes<16 hold K = 0..15,
// lanes>=16 hold K = 16..31, in order (2 per VGPR).
DEV v16bf load_a_f32(const float* base, int ld, int m0, int k0, int lane) {
  const float* row = base + (size_t)(m0 + (lane & 15)) * ld + k0 + ((lane & 16) ? 16 : 0);
  v16bf a;
#pragma unroll
  for (int i = 0; i < 8; ++i) a[i] = (__bf16)row[i];
#pragma unroll
  for (int i = 0; i < 8; ++i) a[i + 8] = (__bf16)row[8 + i];
  return a;
}
DEV v16bf load_a_bf16(const __bf16* base, int ld, int m0, int k0, int lane) {
  const __bf16* row = base + (size_t)(m0 + (lane & 15)) * ld + k0 + ((lane & 16) ? 16 : 0);
  v16bf a;
#pragma unroll
  for (int i = 0; i < 8; ++i) a[i] = row[i];
#pragma unroll
  for (int i = 0; i < 8; ++i) a[i + 8] = row[8 + i];
  return a;
}
// B (K=32 x 16): lane column N = lane&15; lanes<16 hold K=0..15,
// lanes>=16 hold K=16..31, in order; source is row-major [N][K].
DEV v16bf load_b_f32(const float* base, int ld, int n0, int k0, int lane) {
  const float* row = base + (size_t)(n0 + (lane & 15)) * ld + k0 + ((lane & 16) ? 16 : 0);
  v16bf b;
#pragma unroll
  for (int i = 0; i < 16; ++i) b[i] = (__bf16)row[i];
  return b;
}
DEV v16bf load_b_bf16(const __bf16* base, int ld, int n0, int k0, int lane) {
  const __bf16* row = base + (size_t)(n0 + (lane & 15)) * ld + k0 + ((lane & 16) ? 16 : 0);
  v16bf b;
#pragma unroll
  for (int i = 0; i < 16; ++i) b[i] = row[i];
  return b;
}

DEV v8f wmma_bf16(v16bf a, v16bf b, v8f c) {
  return __builtin_amdgcn_wmma_f32_16x16x32_bf16(false, a, false, b, (short)0, c,
                                                 false, false);
}

// ---------------- Projection GEMM:  Y[row,e] = sum_d X[row,d] * W[e,d] ------
// MODE 0: Y fp32 row-major [M,N]
// MODE 1: Y bf16 [B,H,S,dk]          (Q, K pre-RoPE)
// MODE 2: Y bf16 [B,H,dk,S]          (V transposed for PV fragment loads)
template <int MODE>
__global__ void __launch_bounds__(256)
proj_gemm(const float* __restrict__ X, const float* __restrict__ W,
          float* __restrict__ Yf, __bf16* __restrict__ Yb) {
  const int lane = threadIdx.x & 31;
  const int wave = blockIdx.x * (blockDim.x >> 5) + (threadIdx.x >> 5);
  const int N = Dd, K = Dd;
  const int ntiles = N / 64;              // 16
  const int m0 = (wave / ntiles) * 16;
  const int n0 = (wave % ntiles) * 64;

  v8f acc[4];
#pragma unroll
  for (int j = 0; j < 4; ++j) acc[j] = 0.0f;

  for (int k0 = 0; k0 < K; k0 += 32) {
    // hoist all loads so they issue as one clause before the WMMA chain
    const v16bf a = load_a_f32(X, K, m0, k0, lane);
    v16bf b[4];
#pragma unroll
    for (int j = 0; j < 4; ++j) b[j] = load_b_f32(W, K, n0 + 16 * j, k0, lane);
#pragma unroll
    for (int j = 0; j < 4; ++j) acc[j] = wmma_bf16(a, b[j], acc[j]);
  }

  const int rlo = (lane & 16) ? 8 : 0;
  const int nl = lane & 15;
#pragma unroll
  for (int j = 0; j < 4; ++j) {
#pragma unroll
    for (int r = 0; r < 8; ++r) {
      const int row = m0 + r + rlo;
      const int col = n0 + 16 * j + nl;
      const float v = acc[j][r];
      if (MODE == 0) {
        Yf[(size_t)row * N + col] = v;
      } else {
        const int b2 = row / Ss, s = row % Ss;
        const int h = col / DK, dk = col % DK;
        if (MODE == 1)
          Yb[(((size_t)(b2 * Hh + h) * Ss) + s) * DK + dk] = (__bf16)v;
        else
          Yb[(((size_t)(b2 * Hh + h) * DK) + dk) * Ss + s] = (__bf16)v;
      }
    }
  }
}

// ---------------- In-place RoPE (+optional exact pow2 scale) on bf16 --------
__global__ void __launch_bounds__(256)
rope_inplace(__bf16* __restrict__ Q, int total, float scale) {
  const int idx = blockIdx.x * blockDim.x + threadIdx.x;
  if (idx >= total) return;
  const int j = idx & 31;                 // pair index 0..31
  const int s = (idx >> 5) & (Ss - 1);
  const int bh = idx >> 5 >> 11;          // idx / (32*Ss)
  __bf16* p = Q + ((size_t)bh * Ss + s) * DK + 2 * j;
  const float freq = __powf(10000.0f, -(float)(2 * j) / (float)DK);
  const float ang = (float)s * freq;
  const float cs = __cosf(ang) * scale, sn = __sinf(ang) * scale;
  const float x1 = (float)p[0], x2 = (float)p[1];
  p[0] = (__bf16)(cs * x1 - sn * x2);
  p[1] = (__bf16)(sn * x1 + cs * x2);
}

// ---------------- Flash attention: 1 wave per (b,h, 16-query tile) ----------
// Q is pre-scaled by 1/sqrt(dk) (exact, power of two), so scores need no mul.
__global__ void __launch_bounds__(32)
flash_attn(const __bf16* __restrict__ Qb, const __bf16* __restrict__ Kb,
           const __bf16* __restrict__ Vt, float* __restrict__ attn) {
  __shared__ float  S_lds[16][32];
  __shared__ __bf16 P_lds[16][32];
  __shared__ float  alpha_lds[16];
  __shared__ float  l_lds[16];

  const int lane = threadIdx.x;           // blockDim.x == 32
  const int qtiles = Ss / 16;             // 128
  const int qt = blockIdx.x % qtiles;
  const int bh = blockIdx.x / qtiles;     // 0..31
  const int q0 = qt * 16;

  const __bf16* Qp = Qb + (size_t)bh * Ss * DK;
  const __bf16* Kp = Kb + (size_t)bh * Ss * DK;
  const __bf16* Vp = Vt + (size_t)bh * DK * Ss;

  const v16bf aq0 = load_a_bf16(Qp, DK, q0, 0, lane);
  const v16bf aq1 = load_a_bf16(Qp, DK, q0, 32, lane);

  v8f o[4];
#pragma unroll
  for (int j = 0; j < 4; ++j) o[j] = 0.0f;
  float m_i = -3.0e38f, l_i = 0.0f;

  const int rlo = (lane & 16) ? 8 : 0;
  const int nl = lane & 15;
  const int ktmax = (q0 + 15) / 32;

  for (int kt = 0; kt <= ktmax; ++kt) {
    const int key0 = kt * 32;

    // prefetch next key tile (global_prefetch_b8)
    if (kt < ktmax) {
      const int nk = key0 + 32;
      __builtin_prefetch(Kp + (size_t)(nk + lane) * DK, 0, 3);
      __builtin_prefetch(Vp + (size_t)lane * Ss + nk, 0, 3);
      __builtin_prefetch(Vp + (size_t)(lane + 32) * Ss + nk, 0, 3);
    }

    // ---- scores S = Q K^T (Q pre-scaled), causal-masked, staged to LDS ----
    // Hoist only the 4 K-fragments: consumed immediately, no spill pressure.
    {
      v16bf bk[2][2];
#pragma unroll
      for (int nc = 0; nc < 2; ++nc) {
        bk[nc][0] = load_b_bf16(Kp, DK, key0 + 16 * nc, 0, lane);
        bk[nc][1] = load_b_bf16(Kp, DK, key0 + 16 * nc, 32, lane);
      }
#pragma unroll
      for (int nc = 0; nc < 2; ++nc) {
        v8f s = 0.0f;
        s = wmma_bf16(aq0, bk[nc][0], s);
        s = wmma_bf16(aq1, bk[nc][1], s);
#pragma unroll
        for (int r = 0; r < 8; ++r) {
          const int row = r + rlo;
          const int kpos = key0 + 16 * nc + nl;
          S_lds[row][16 * nc + nl] = (kpos <= q0 + row) ? s[r] : -1.0e30f;
        }
      }
    }
    __syncthreads();

    // ---- online softmax: lanes 0..15 each own one query row ----
    if (lane < 16) {
      float rmax = -3.0e38f;
#pragma unroll
      for (int c = 0; c < 32; ++c) rmax = fmaxf(rmax, S_lds[lane][c]);
      const float new_m = fmaxf(m_i, rmax);
      const float alpha = __expf(m_i - new_m);
      float tsum = 0.0f;
#pragma unroll
      for (int c = 0; c < 32; ++c) {
        const float pv = __expf(S_lds[lane][c] - new_m);
        tsum += pv;
        P_lds[lane][c] = (__bf16)pv;
      }
      m_i = new_m;
      l_i = l_i * alpha + tsum;
      alpha_lds[lane] = alpha;
    }
    __syncthreads();

    // ---- O = O*alpha + P V ----
    // V fragments loaded HERE (short liveness): one 8x b128 clause, consumed
    // immediately by the PV WMMAs while alpha rescale overlaps.
    {
      v16bf bv[4];
#pragma unroll
      for (int j = 0; j < 4; ++j) bv[j] = load_b_bf16(Vp, Ss, 16 * j, key0, lane);
      const v16bf pa = load_a_bf16(&P_lds[0][0], 32, 0, 0, lane);
#pragma unroll
      for (int j = 0; j < 4; ++j) {
#pragma unroll
        for (int r = 0; r < 8; ++r) o[j][r] *= alpha_lds[r + rlo];
        o[j] = wmma_bf16(pa, bv[j], o[j]);
      }
    }
    __syncthreads();
  }

  if (lane < 16) l_lds[lane] = l_i;
  __syncthreads();

  const int b = bh / Hh, h = bh % Hh;
#pragma unroll
  for (int j = 0; j < 4; ++j) {
#pragma unroll
    for (int r = 0; r < 8; ++r) {
      const int row = r + rlo;
      attn[((size_t)b * Ss + q0 + row) * Dd + h * DK + 16 * j + nl] =
          o[j][r] / l_lds[row];
    }
  }
}

// ---------------- Host launch ------------------------------------------------
extern "C" void kernel_launch(void* const* d_in, const int* in_sizes, int n_in,
                              void* d_out, int out_size, void* d_ws, size_t ws_size,
                              hipStream_t stream) {
  (void)in_sizes; (void)n_in; (void)out_size; (void)ws_size;
  const float* x  = (const float*)d_in[0];
  const float* Wq = (const float*)d_in[1];
  const float* Wk = (const float*)d_in[2];
  const float* Wv = (const float*)d_in[3];
  const float* Wo = (const float*)d_in[4];
  float* out = (float*)d_out;

  char* ws = (char*)d_ws;
  const size_t MB = 1024ull * 1024ull;
  __bf16* qb = (__bf16*)(ws);              //  8 MB  [B,H,S,dk]
  __bf16* kb = (__bf16*)(ws + 8 * MB);     //  8 MB  [B,H,S,dk]
  __bf16* vt = (__bf16*)(ws + 16 * MB);    //  8 MB  [B,H,dk,S]
  float*  at = (float*) (ws + 24 * MB);    // 16 MB  [B,S,D]

  const int waves = (Bb * Ss / 16) * (Dd / 64);  // 4096 waves
  dim3 blk(256);
  dim3 grd(waves / 8);                           // 8 waves/block -> 512 blocks

  proj_gemm<1><<<grd, blk, 0, stream>>>(x, Wq, nullptr, qb);
  proj_gemm<1><<<grd, blk, 0, stream>>>(x, Wk, nullptr, kb);
  proj_gemm<2><<<grd, blk, 0, stream>>>(x, Wv, nullptr, vt);

  const int rope_total = Bb * Hh * Ss * 32;
  rope_inplace<<<rope_total / 256, 256, 0, stream>>>(qb, rope_total, ATTN_SCALE);
  rope_inplace<<<rope_total / 256, 256, 0, stream>>>(kb, rope_total, 1.0f);

  flash_attn<<<Bb * Hh * (Ss / 16), 32, 0, stream>>>(qb, kb, vt, at);

  proj_gemm<0><<<grd, blk, 0, stream>>>(at, Wo, out, nullptr);
}
